// MinibatchDiscrimination_3831110828622
// MI455X (gfx1250) — matrix operand order repaired and checked by reading, not compiled, not run on hardware
//
#include <hip/hip_runtime.h>

typedef __attribute__((ext_vector_type(16))) _Float16 v16h;
typedef __attribute__((ext_vector_type(8)))  _Float16 v8h;
typedef __attribute__((ext_vector_type(8)))  float    v8f;
typedef __attribute__((ext_vector_type(2)))  _Float16 h2;

#define BATCH 512
#define DIM   256
#define NKER  64
#define KDIM  16
#define OUTW  (DIM + NKER)    // 320
#define SWT_PITCH (DIM + 8)   // 264 halves: +8 pad staggers LDS banks per column

// ---------------------------------------------------------------------------
// Kernel 1: act[k][i][m] = sum_d x[i][d] * W[k][d][m] via v_wmma_f32_16x16x32_f16.
// One 1024-thread block (32 waves) per kernel k; wave w owns row-tile w.
// W[k] is staged ONCE per block into LDS, f16 + transposed (col-major, padded
// pitch), so each wave's B fragment is two aligned 16B LDS reads per K-step.
// grid = NKER, block = 1024.
// ---------------------------------------------------------------------------
__global__ void __launch_bounds__(1024)
act_wmma_kernel(const float* __restrict__ x,      // (BATCH, DIM)
                const float* __restrict__ W,      // (NKER, DIM, KDIM)
                _Float16*    __restrict__ act)    // (NKER, BATCH, KDIM) f16
{
    __shared__ __align__(16) _Float16 sWt[KDIM * SWT_PITCH];   // 8.25 KB

    const int k = blockIdx.x;
    const float* __restrict__ Wk = W + (size_t)k * DIM * KDIM;

    // ---- stage W[k] -> LDS, f32->f16, transposed to (col, d)
    for (int t = threadIdx.x; t < DIM * KDIM; t += blockDim.x) {
        const int d = t >> 4, col = t & 15;
        sWt[col * SWT_PITCH + d] = (_Float16)Wk[t];
    }
    __syncthreads();

    const int wave  = threadIdx.x >> 5;            // 0..31 == row tile
    const int lane  = threadIdx.x & 31;
    const int lhalf = lane >> 4;                   // 0 or 1
    const int lcol  = lane & 15;

    const float*    __restrict__ xrow = x + (size_t)(wave * 16 + lcol) * DIM;
    const _Float16* __restrict__ bcol = sWt + lcol * SWT_PITCH;

    v8f c = {};
    for (int kk = 0; kk < DIM; kk += 32) {
        // ---- A fragment (16x32 f16): two contiguous 8-float runs per lane
        const int ka = kk + lhalf * 8;
        v16h a;
#pragma unroll
        for (int h = 0; h < 8; ++h) {
            a[h]     = (_Float16)xrow[ka + h];        // K = ka + h
            a[h + 8] = (_Float16)xrow[ka + 16 + h];   // K = ka + 16 + h
        }
        // ---- B fragment (32x16 f16): column lcol, rows kb..kb+7 and kb+16..kb+23, from LDS
        const int kb = kk + lhalf * 8;
        const v8h blo = *(const v8h*)(bcol + kb);        // 16B aligned
        const v8h bhi = *(const v8h*)(bcol + kb + 16);   // 16B aligned
        const v16h bf = __builtin_shufflevector(blo, bhi,
                0, 1, 2, 3, 4, 5, 6, 7, 8, 9, 10, 11, 12, 13, 14, 15);

        c = __builtin_amdgcn_wmma_f32_16x16x32_f16(
                /*neg_a=*/false, a, /*neg_b=*/false, bf,
                /*c_mod=*/(short)0, c, /*reuse_a=*/false, /*reuse_b=*/false);
    }

    // ---- Store D: lane l, vgpr r holds (M = r + 8*lhalf, N = lcol)
    _Float16* __restrict__ actk = act + ((size_t)k * BATCH + wave * 16) * KDIM;
#pragma unroll
    for (int r = 0; r < 8; ++r) {
        actk[(size_t)(r + 8 * lhalf) * KDIM + lcol] = (_Float16)c[r];
    }
}

// ---------------------------------------------------------------------------
// Kernel 2: feats[i][k] = b[k] + sum_j exp(-||act[i,k,:]-act[j,k,:]||_1)
// One block per kernel k. The 512x16 f16 slice (16 KB) is staged into LDS
// with GLOBAL_LOAD_ASYNC_TO_LDS_B128 (no VGPR round-trip, ASYNCcnt-tracked),
// then the O(B^2) L1 loop runs on packed f16 (abs = &0x7FFF7FFF).
// grid = NKER, block = 256.
// ---------------------------------------------------------------------------
__device__ __forceinline__ h2 habs2(h2 v) {
    unsigned u = __builtin_bit_cast(unsigned, v) & 0x7FFF7FFFu;
    return __builtin_bit_cast(h2, u);
}

__global__ void __launch_bounds__(256)
feats_kernel(const _Float16* __restrict__ act,   // (NKER, BATCH, KDIM) f16
             const float*    __restrict__ b,     // (NKER,)
             float*          __restrict__ out)   // (BATCH, OUTW)
{
    __shared__ __align__(16) _Float16 sact[BATCH * KDIM];   // 16 KB

    const int k = blockIdx.x;
    const _Float16* __restrict__ actk = act + (size_t)k * BATCH * KDIM;

    // ---- async global -> LDS staging: 1024 x 16B, 4 per thread
    {
        const unsigned lds_base = (unsigned)(uintptr_t)(void*)sact;  // LDS byte offset
        for (int t = threadIdx.x; t < (BATCH * KDIM) / 8; t += blockDim.x) {
            const unsigned goff = (unsigned)t * 16u;
            const unsigned loff = lds_base + (unsigned)t * 16u;
            asm volatile("global_load_async_to_lds_b128 %0, %1, %2"
                         :
                         : "v"(loff), "v"(goff), "s"(actk)
                         : "memory");
        }
        asm volatile("s_wait_asynccnt 0" ::: "memory");
    }
    __syncthreads();

    const float bk = b[k];

    for (int i = threadIdx.x; i < BATCH; i += blockDim.x) {
        const h2* __restrict__ ai = (const h2*)(sact + i * KDIM);
        h2 a0 = ai[0], a1 = ai[1], a2 = ai[2], a3 = ai[3];
        h2 a4 = ai[4], a5 = ai[5], a6 = ai[6], a7 = ai[7];

        float acc = 0.0f;
        for (int j = 0; j < BATCH; ++j) {
            const h2* __restrict__ aj = (const h2*)(sact + j * KDIM);  // LDS broadcast
            h2 s;
            s  = habs2(a0 - aj[0]);
            s += habs2(a1 - aj[1]);
            s += habs2(a2 - aj[2]);
            s += habs2(a3 - aj[3]);
            s += habs2(a4 - aj[4]);
            s += habs2(a5 - aj[5]);
            s += habs2(a6 - aj[6]);
            s += habs2(a7 - aj[7]);
            const float l1 = (float)s[0] + (float)s[1];
            acc += __expf(-l1);
        }
        out[(size_t)i * OUTW + DIM + k] = acc + bk;
    }
}

// ---------------------------------------------------------------------------
// Kernel 3: out[:, 0:256] = x   (bandwidth-trivial copy)
// ---------------------------------------------------------------------------
__global__ void __launch_bounds__(64)
copy_x_kernel(const float* __restrict__ x, float* __restrict__ out)
{
    const int i = blockIdx.x;
    const float4* src = (const float4*)(x + (size_t)i * DIM);
    float4*       dst = (float4*)(out + (size_t)i * OUTW);
    dst[threadIdx.x] = src[threadIdx.x];
}

// ---------------------------------------------------------------------------
extern "C" void kernel_launch(void* const* d_in, const int* in_sizes, int n_in,
                              void* d_out, int out_size, void* d_ws, size_t ws_size,
                              hipStream_t stream) {
    const float* x = (const float*)d_in[0];   // (512, 256)
    const float* W = (const float*)d_in[1];   // (64, 256, 16)
    const float* b = (const float*)d_in[2];   // (64,)
    float* out     = (float*)d_out;           // (512, 320)
    _Float16* act  = (_Float16*)d_ws;         // (64, 512, 16) f16 = 1 MB scratch

    act_wmma_kernel<<<NKER, 1024, 0, stream>>>(x, W, act);
    feats_kernel<<<NKER, 256, 0, stream>>>(act, b, out);
    copy_x_kernel<<<BATCH, 64, 0, stream>>>(x, out);
}
